// MultiHeadAttention_1529008357713
// MI455X (gfx1250) — matrix operand order repaired and checked by reading, not compiled, not run on hardware
//
#include <hip/hip_runtime.h>
#include <hip/hip_bf16.h>

// ---------------------------------------------------------------------------
// Problem constants (from reference)
// ---------------------------------------------------------------------------
#define B_   2
#define L_   1024
#define D_   1024
#define H_   8
#define DV_  1536
#define HDV_ (H_ * DV_)   // 12288

typedef __bf16 bf16_t;
typedef __attribute__((ext_vector_type(16))) __bf16 bf16x16;
typedef __attribute__((ext_vector_type(8)))  __bf16 bf16x8;
typedef __attribute__((ext_vector_type(4)))  __bf16 bf16x4;
typedef __attribute__((ext_vector_type(8)))  float  f32x8;

// ---------------------------------------------------------------------------
// Tiled bf16 WMMA GEMM:  C[m,n] = sum_k A[m,k] * B[n,k]  (+ bias[n])
// A, B fp32 in memory, converted to bf16 at LDS staging; fp32 accumulate.
// Tile: BM=64 x BN=128 x BK=32, 256 threads = 8 waves (2 x 4), each wave 32x32.
// Requires: K % 32 == 0, N % 128 == 0 (true for all uses). M guarded.
// Batched via blockIdx.z with per-matrix element offsets.
// ---------------------------------------------------------------------------
#define BM 64
#define BN 128
#define BK 32
#define LDSS 40   // BK + 8 pad (80 B row stride, 16 B aligned)

__global__ __launch_bounds__(256)
void wmma_gemm_nt(const float* __restrict__ A,  long long aOff, int lda,
                  const float* __restrict__ Bm, long long bOff, int ldb,
                  float*       __restrict__ C,  long long cOff, int ldc,
                  const float* __restrict__ bias, int M, int K)
{
    __shared__ bf16_t As[BM][LDSS];
    __shared__ bf16_t Bs[BN][LDSS];

    const int z = blockIdx.z;
    A  += (long long)z * aOff;
    Bm += (long long)z * bOff;
    C  += (long long)z * cOff;

    const int tid  = threadIdx.x;
    const int lane = tid & 31;
    const int wave = tid >> 5;     // 0..7
    const int wrow = wave >> 2;    // 0..1 : 32 rows each
    const int wcol = wave & 3;     // 0..3 : 32 cols each
    const int half = lane >> 4;    // lane group (K split per ISA layout)
    const int lr   = lane & 15;

    const int mTile = blockIdx.y * BM;
    const int nTile = blockIdx.x * BN;

    f32x8 acc[2][2] = {};

    const int kTiles = K / BK;
    for (int kt = 0; kt < kTiles; ++kt) {
        const int k0 = kt * BK;
        __syncthreads();
        // ---- stage A tile (64x32), fp32 -> bf16, 2 float4 groups / thread
        #pragma unroll
        for (int i = 0; i < 2; ++i) {
            int g  = tid + i * 256;
            int r  = g >> 3;
            int c4 = (g & 7) * 4;
            int gr = mTile + r;
            float4 f;
            if (gr < M) f = *(const float4*)(A + (long long)gr * lda + k0 + c4);
            else        f = make_float4(0.f, 0.f, 0.f, 0.f);
            bf16x4 hv = { (bf16_t)f.x, (bf16_t)f.y, (bf16_t)f.z, (bf16_t)f.w };
            *(bf16x4*)(&As[r][c4]) = hv;
        }
        // ---- stage B tile (128x32, [n][k]), 4 float4 groups / thread
        #pragma unroll
        for (int i = 0; i < 4; ++i) {
            int g  = tid + i * 256;
            int r  = g >> 3;
            int c4 = (g & 7) * 4;
            float4 f = *(const float4*)(Bm + (long long)(nTile + r) * ldb + k0 + c4);
            bf16x4 hv = { (bf16_t)f.x, (bf16_t)f.y, (bf16_t)f.z, (bf16_t)f.w };
            *(bf16x4*)(&Bs[r][c4]) = hv;
        }
        __syncthreads();

        // ---- A fragments: 16x32, lanes 0-15 hold M=row,K 0..7/16..23; lanes
        //      16-31 hold K 8..15/24..31  -> two 16B LDS reads per fragment.
        bf16x16 afr[2], bfr[2];
        #pragma unroll
        for (int mi = 0; mi < 2; ++mi) {
            int r = wrow * 32 + mi * 16 + lr;
            bf16x8 lo = *(const bf16x8*)(&As[r][half * 8]);
            bf16x8 hi = *(const bf16x8*)(&As[r][half * 8 + 16]);
            afr[mi] = __builtin_shufflevector(lo, hi, 0,1,2,3,4,5,6,7,8,9,10,11,12,13,14,15);
        }
        // ---- B fragments: 32x16, lanes 0-15 col=n K=0..7/16..23; lanes 16-31 K=8..15/24..31
        #pragma unroll
        for (int ni = 0; ni < 2; ++ni) {
            int r = wcol * 32 + ni * 16 + lr;
            bf16x8 lo = *(const bf16x8*)(&Bs[r][half * 8]);
            bf16x8 hi = *(const bf16x8*)(&Bs[r][half * 8 + 16]);
            bfr[ni] = __builtin_shufflevector(lo, hi, 0,1,2,3,4,5,6,7,8,9,10,11,12,13,14,15);
        }

        #pragma unroll
        for (int mi = 0; mi < 2; ++mi)
            #pragma unroll
            for (int ni = 0; ni < 2; ++ni)
                acc[mi][ni] = __builtin_amdgcn_wmma_f32_16x16x32_bf16(
                    false, afr[mi], false, bfr[ni], (short)0, acc[mi][ni], false, false);
    }

    // ---- epilogue: C layout = VGPR r: row = 8*half + r, col = lr
    #pragma unroll
    for (int mi = 0; mi < 2; ++mi) {
        #pragma unroll
        for (int ni = 0; ni < 2; ++ni) {
            int colG = nTile + wcol * 32 + ni * 16 + lr;
            float bv = bias ? bias[colG] : 0.f;
            #pragma unroll
            for (int r = 0; r < 8; ++r) {
                int rowG = mTile + wrow * 32 + mi * 16 + half * 8 + r;
                if (rowG < M)
                    C[(long long)rowG * ldc + colG] = acc[mi][ni][r] + bv;
            }
        }
    }
}

// ---------------------------------------------------------------------------
// vbar[b,d] = mean_l v[b,l,d] : two-stage column reduction
// ---------------------------------------------------------------------------
__global__ __launch_bounds__(256)
void vbar_partial(const float* __restrict__ v, float* __restrict__ part)
{
    int d = blockIdx.x * 256 + threadIdx.x;
    int b = blockIdx.y;
    int c = blockIdx.z;                      // 16 chunks of 64 rows
    const float* p = v + ((long long)b * L_ + c * 64) * D_ + d;
    float s = 0.f;
    #pragma unroll 4
    for (int l = 0; l < 64; ++l) s += p[(long long)l * D_];
    part[(c * B_ + b) * D_ + d] = s;
}

__global__ __launch_bounds__(256)
void vbar_finish(const float* __restrict__ part, float* __restrict__ vbar)
{
    int d = blockIdx.x * 256 + threadIdx.x;
    int b = blockIdx.y;
    float s = 0.f;
    #pragma unroll
    for (int c = 0; c < 16; ++c) s += part[(c * B_ + b) * D_ + d];
    vbar[b * D_ + d] = s * (1.0f / (float)L_);
}

// ---------------------------------------------------------------------------
// Geven[b,d] = sum_{h even} G[h][b][d] ; Godd likewise. G is [H][B][D].
// ---------------------------------------------------------------------------
__global__ __launch_bounds__(256)
void reduce_heads(const float* __restrict__ G,
                  float* __restrict__ Geven, float* __restrict__ Godd)
{
    int i = blockIdx.x * 256 + threadIdx.x;   // i in [0, B*D)
    int b = i / D_, d = i % D_;
    float e = 0.f, o = 0.f;
    #pragma unroll
    for (int h = 0; h < H_; h += 2) {
        e += G[((h    ) * B_ + b) * D_ + d];
        o += G[((h + 1) * B_ + b) * D_ + d];
    }
    Geven[i] = e;
    Godd[i]  = o;
}

// ---------------------------------------------------------------------------
// atten[bh,q,k] = mask[(bh % B), q] ? 1/L : 0   (softmax of masked rows
// is exactly uniform; unmasked rows are multiplied by new_mask == 0)
// ---------------------------------------------------------------------------
__global__ __launch_bounds__(256)
void atten_fill(const int* __restrict__ mask, float* __restrict__ atten)
{
    int row = blockIdx.x;              // bh * L + q, 16*1024 rows
    int bh  = row >> 10;
    int qq  = row & (L_ - 1);
    int mv  = mask[(bh & (B_ - 1)) * L_ + qq];
    float val = mv ? (1.0f / (float)L_) : 0.0f;
    float4 f = make_float4(val, val, val, val);
    *(float4*)(atten + (long long)row * L_ + threadIdx.x * 4) = f;
}

// ---------------------------------------------------------------------------
// x = mask[0,l]*Geven[b,:] + mask[1,l]*Godd[b,:] + residual q[b,l,:]
// out = LayerNorm(x) ; one row per block, wave32 shuffle reduction
// ---------------------------------------------------------------------------
__global__ __launch_bounds__(256)
void out_ln(const int* __restrict__ mask, const float* __restrict__ qres,
            const float* __restrict__ Geven, const float* __restrict__ Godd,
            const float* __restrict__ ln_scale, const float* __restrict__ ln_bias,
            float* __restrict__ outp)
{
    __shared__ float sh_s[8], sh_s2[8];
    int bl = blockIdx.x;               // b * L + l
    int b  = bl >> 10;
    int l  = bl & (L_ - 1);
    float m0 = (float)mask[l];         // mask row 0 (heads with even h)
    float m1 = (float)mask[L_ + l];    // mask row 1 (heads with odd h)

    int tid = threadIdx.x;
    float x[4];
    float s = 0.f, s2 = 0.f;
    #pragma unroll
    for (int i = 0; i < 4; ++i) {
        int d = tid + i * 256;
        float xv = m0 * Geven[b * D_ + d] + m1 * Godd[b * D_ + d]
                 + qres[(long long)bl * D_ + d];
        x[i] = xv;
        s += xv;
        s2 += xv * xv;
    }
    #pragma unroll
    for (int off = 16; off > 0; off >>= 1) {
        s  += __shfl_down(s,  off, 32);
        s2 += __shfl_down(s2, off, 32);
    }
    int wave = tid >> 5, lane = tid & 31;
    if (lane == 0) { sh_s[wave] = s; sh_s2[wave] = s2; }
    __syncthreads();
    if (tid == 0) {
        float t = 0.f, t2 = 0.f;
        #pragma unroll
        for (int w = 0; w < 8; ++w) { t += sh_s[w]; t2 += sh_s2[w]; }
        sh_s[0] = t; sh_s2[0] = t2;
    }
    __syncthreads();
    float mu  = sh_s[0] * (1.0f / (float)D_);
    float var = sh_s2[0] * (1.0f / (float)D_) - mu * mu;
    float inv = rsqrtf(var + 1e-5f);
    #pragma unroll
    for (int i = 0; i < 4; ++i) {
        int d = tid + i * 256;
        outp[(long long)bl * D_ + d] = (x[i] - mu) * inv * ln_scale[d] + ln_bias[d];
    }
}

// ---------------------------------------------------------------------------
// Launch
// ---------------------------------------------------------------------------
extern "C" void kernel_launch(void* const* d_in, const int* in_sizes, int n_in,
                              void* d_out, int out_size, void* d_ws, size_t ws_size,
                              hipStream_t stream)
{
    const float* q_in = (const float*)d_in[0];   // (B,L,D)  residual
    const float* v_in = (const float*)d_in[2];   // (B,L,D)
    const int*   mask = (const int*)  d_in[3];   // (B,L,1) int32
    const float* wv   = (const float*)d_in[8];   // (HDV, D)
    const float* bv   = (const float*)d_in[9];   // (HDV,)
    const float* fcw  = (const float*)d_in[10];  // (D, HDV)
    const float* lns  = (const float*)d_in[11];  // (D,)
    const float* lnb  = (const float*)d_in[12];  // (D,)

    float* out   = (float*)d_out;                     // (B,L,D)
    float* atten = out + (size_t)B_ * L_ * D_;        // (B*H, L, L)

    // workspace carve-out (floats)
    float* wsf   = (float*)d_ws;
    float* part  = wsf;                   // 16 * B * D
    float* vbar  = part + 16 * B_ * D_;   // B * D
    float* vmp   = vbar + B_ * D_;        // B * HDV   (mean attention output per head)
    float* G     = vmp + B_ * HDV_;       // H * B * D
    float* Geven = G + H_ * B_ * D_;      // B * D
    float* Godd  = Geven + B_ * D_;       // B * D

    // 1) vbar = mean over L of v
    vbar_partial<<<dim3(D_ / 256, B_, 16), 256, 0, stream>>>(v_in, part);
    vbar_finish <<<dim3(D_ / 256, B_),     256, 0, stream>>>(part, vbar);

    // 2) vmp[b, n] = vbar[b,:] . w_vs[n,:] + b_vs[n]   (M=2, K=1024, N=12288)
    wmma_gemm_nt<<<dim3(HDV_ / BN, 1, 1), 256, 0, stream>>>(
        vbar, 0LL, D_, wv, 0LL, D_, vmp, 0LL, HDV_, bv, B_, D_);

    // 3) G[h][b][d] = sum_dv vmp[b, h*DV+dv] * fc_w[d, h*DV+dv]
    //    batched over h: M=2, K=1536, N=1024
    wmma_gemm_nt<<<dim3(D_ / BN, 1, H_), 256, 0, stream>>>(
        vmp, (long long)DV_, HDV_, fcw, (long long)DV_, HDV_,
        G, (long long)(B_ * D_), D_, nullptr, B_, DV_);

    // 4) fold heads by mask parity
    reduce_heads<<<dim3(B_ * D_ / 256), 256, 0, stream>>>(G, Geven, Godd);

    // 5) atten output (uniform-or-zero rows)
    atten_fill<<<dim3(B_ * H_ * L_), 256, 0, stream>>>(mask, atten);

    // 6) residual + LayerNorm
    out_ln<<<dim3(B_ * L_), 256, 0, stream>>>(mask, q_in, Geven, Godd, lns, lnb, out);
}
